// MultiHeadAtt_69277822485127
// MI455X (gfx1250) — matrix operand order repaired and checked by reading, not compiled, not run on hardware
//
#include <hip/hip_runtime.h>
#include <hip/hip_bf16.h>

// ---------------- problem constants ----------------
constexpr int Bc = 4;
constexpr int Sc = 2048;
constexpr int Ec = 1024;
constexpr int Hc = 512;            // per-head dim, 2 heads

// f16 workspace layout (element offsets)
constexpr size_t PLANE  = (size_t)Bc * 2 * Sc * Hc;      // 8,388,608 halves
constexpr size_t XH_OFF = 0;
constexpr size_t Q_OFF  = 1 * PLANE;
constexpr size_t K_OFF  = 2 * PLANE;
constexpr size_t VT_OFF = 3 * PLANE;                     // V transposed [B][2][H][S]
constexpr size_t AT_OFF = 4 * PLANE;                     // attention out [B][2][S][H]
constexpr size_t W_ELE  = (size_t)2 * Hc * Hc;           // 524,288
constexpr size_t WQ_OFF = 5 * PLANE;
constexpr size_t WK_OFF = WQ_OFF + W_ELE;
constexpr size_t WV_OFF = WK_OFF + W_ELE;
constexpr size_t WF_OFF = WV_OFF + W_ELE;                // [E][E] = 1,048,576

// ---------------- WMMA helpers (wave32, gfx1250) ----------------
typedef _Float16 v16h __attribute__((ext_vector_type(16)));
typedef _Float16 v8h  __attribute__((ext_vector_type(8)));
typedef float    v8f  __attribute__((ext_vector_type(8)));
typedef unsigned int u32x4 __attribute__((ext_vector_type(4)));
typedef int          i32x4 __attribute__((ext_vector_type(4)));
typedef int          i32x8 __attribute__((ext_vector_type(8)));

__device__ __forceinline__ v16h ld_pair(const _Float16* p0, const _Float16* p1) {
  union U { v16h v; v8h h[2]; } u;
  u.h[0] = *(const v8h*)p0;
  u.h[1] = *(const v8h*)p1;
  return u.v;
}
// A fragment (16x32 f16): A row-major, `a` points at tile origin (row0,k0), lda in halves.
__device__ __forceinline__ v16h ld_afrag(const _Float16* a, int lda, int lane) {
  const int r = lane & 15, g = lane >> 4;
  const _Float16* base = a + (size_t)r * lda + g * 8;
  return ld_pair(base, base + 16);
}
// B fragment (32x16 f16): stored as Bt row-major [N][K]; `bt` at (n0,k0), ldb in halves.
__device__ __forceinline__ v16h ld_bfrag(const _Float16* bt, int ldb, int lane) {
  const int c = lane & 15, g = lane >> 4;
  const _Float16* base = bt + (size_t)c * ldb + g * 8;
  return ld_pair(base, base + 16);
}
__device__ __forceinline__ v8f wmma16(v16h a, v16h b, v8f c) {
  return __builtin_amdgcn_wmma_f32_16x16x32_f16(false, a, false, b, (short)0, c,
                                                false, false);
}

// ---------------- TDM: async 2D f16 tile load -> LDS ----------------
// Loads tileN rows x tileK halves from a row-major f16 tensor (row stride tensK
// halves) into LDS at lds_off, inserting 4 DWORDs (8 halves) of padding after
// every 64 DWORDs (128 halves) -> LDS row stride = tileK + 8 halves.
// D# layout per CDNA5 ISA ch.8 (group0 / group1; 2D => groups 2,3 zero).
__device__ __forceinline__ void tdm_load_2d(unsigned lds_off, const _Float16* gptr,
                                            int tensK, int tensN, int tileK,
                                            int tileN) {
  unsigned long long ga = (unsigned long long)(uintptr_t)gptr;
  u32x4 g0;
  g0[0] = 1u;                                        // count=1 valid descriptor
  g0[1] = lds_off;                                   // LDS byte address
  g0[2] = (unsigned)(ga & 0xffffffffu);              // global_addr[31:0]
  g0[3] = (unsigned)((ga >> 32) & 0x01ffffffu)       // global_addr[56:32]
        | (2u << 30);                                // type = 2 ("image")
  i32x8 g1;
  g1[0] = (1 << 16)                                  // data_size = 2 bytes
        | (1 << 20)                                  // pad_enable
        | (5 << 22)                                  // pad_interval: 64 DWORDs
        | (3 << 25);                                 // pad_amount: 4 DWORDs
  g1[1] = (tensK & 0xffff) << 16;                    // tensor_dim0[15:0]
  g1[2] = ((tensK >> 16) & 0xffff) | ((tensN & 0xffff) << 16);
  g1[3] = ((tensN >> 16) & 0xffff) | ((tileK & 0xffff) << 16); // tile_dim0
  g1[4] = (tileN & 0xffff);                          // tile_dim1 (tile_dim2=0)
  g1[5] = tensK;                                     // tensor_dim0_stride lo
  g1[6] = 0;
  g1[7] = 0;
  i32x4 z4 = {0, 0, 0, 0};
#if defined(__clang_major__) && (__clang_major__ >= 23)
  i32x8 z8 = {0, 0, 0, 0, 0, 0, 0, 0};
  __builtin_amdgcn_tensor_load_to_lds(g0, g1, z4, z4, z8, 0);
#else
  __builtin_amdgcn_tensor_load_to_lds(g0, g1, z4, z4, 0);
#endif
}

// ---------------- kernel 1: casts ----------------
__global__ void cast_f16(const float* __restrict__ src, _Float16* __restrict__ dst, int n) {
  int i = blockIdx.x * blockDim.x + threadIdx.x;
  if (i < n) dst[i] = (_Float16)src[i];
}

// input [B][S][E] fp32 -> xh [B][2][S][H] f16
__global__ void remap_input(const float* __restrict__ in, _Float16* __restrict__ xh) {
  int i = blockIdx.x * blockDim.x + threadIdx.x;          // < B*2*S*H = 2^23
  if (i >= (int)PLANE) return;
  int h  = i & (Hc - 1);
  int s  = (i >> 9) & (Sc - 1);
  int bn = i >> 20;                                        // /(S*H)
  int b  = bn >> 1, n = bn & 1;
  xh[i] = (_Float16)in[((size_t)(b * Sc + s)) * Ec + n * Hc + h];
}

// ---------------- kernel 2: QKV projections ----------------
// grid.x = 1024 (8 waves/block, 8192 tiles), grid.y = 3 (q,k,v)
__global__ __launch_bounds__(256) void qkv_gemm(
    const _Float16* __restrict__ xh,
    const _Float16* __restrict__ wq, const _Float16* __restrict__ wk,
    const _Float16* __restrict__ wv,
    const float* __restrict__ bq, const float* __restrict__ bk,
    const float* __restrict__ bv,
    _Float16* __restrict__ Qo, _Float16* __restrict__ Ko, _Float16* __restrict__ Vto) {
  const int lane = threadIdx.x & 31;
  const int w    = threadIdx.x >> 5;
  const int z    = blockIdx.y;
  const _Float16* W  = (z == 0) ? wq : (z == 1) ? wk : wv;
  const float* bias  = (z == 0) ? bq : (z == 1) ? bk : bv;

  const int gt = blockIdx.x * 8 + w;   // 0..8191
  const int bn = gt >> 10;             // (b,head) plane 0..7
  const int r  = gt & 1023;
  const int rb = r >> 3;               // row block 0..127
  const int cg = r & 7;                // 64-col group 0..7
  const int nh = bn & 1;

  const _Float16* A  = xh + (size_t)bn * Sc * Hc + (size_t)rb * 16 * Hc;
  const _Float16* Wh = W + (size_t)nh * Hc * Hc;
  const int o0 = cg * 64;
  const int cn = lane & 15, g = lane >> 4;

  v8f acc[4];
#pragma unroll
  for (int j = 0; j < 4; ++j) {
    float bval = bias[nh * Hc + o0 + j * 16 + cn];
#pragma unroll
    for (int v = 0; v < 8; ++v) acc[j][v] = bval;
  }
  for (int ks = 0; ks < Hc / 32; ++ks) {
    v16h a = ld_afrag(A + ks * 32, Hc, lane);
#pragma unroll
    for (int j = 0; j < 4; ++j) {
      v16h b = ld_bfrag(Wh + (size_t)(o0 + j * 16) * Hc + ks * 32, Hc, lane);
      acc[j] = wmma16(a, b, acc[j]);
    }
  }
  const int s0 = rb * 16;
  if (z < 2) {
    _Float16* O = ((z == 0) ? Qo : Ko) + (size_t)bn * Sc * Hc;
#pragma unroll
    for (int j = 0; j < 4; ++j)
#pragma unroll
      for (int v = 0; v < 8; ++v)
        O[(size_t)(s0 + v + 8 * g) * Hc + o0 + j * 16 + cn] = (_Float16)acc[j][v];
  } else {
    // V transposed: Vt[h][t]; lane's 8 rows (t) are consecutive -> one b128 store
    _Float16* O = Vto + (size_t)bn * Hc * Sc;
#pragma unroll
    for (int j = 0; j < 4; ++j) {
      v8h pk;
#pragma unroll
      for (int v = 0; v < 8; ++v) pk[v] = (_Float16)acc[j][v];
      *(v8h*)(O + (size_t)(o0 + j * 16 + cn) * Sc + s0 + 8 * g) = pk;
    }
  }
}

// ---------------- kernel 3: flash attention ----------------
// block = 2 waves cooperating on one 16-row Q block; wave w owns k-half (QK^T)
// and h-half (P*V). grid.x = 8 planes * 128 s-blocks = 1024.
__global__ __launch_bounds__(64) void flash_attn(
    const _Float16* __restrict__ Q, const _Float16* __restrict__ K,
    const _Float16* __restrict__ Vt, _Float16* __restrict__ attn,
    const int* __restrict__ maskedp) {
  __shared__ float sx[2][16 * 32];                 // partial score exchange
  __shared__ __align__(16) _Float16 pt[2][16 * 40];// P tile, padded stride 40
  const int lane = threadIdx.x & 31;
  const int w    = threadIdx.x >> 5;
  const int cn   = lane & 15, g = lane >> 4;
  const int msk  = *maskedp;

  const int bn   = blockIdx.x >> 7;
  const int s0   = (blockIdx.x & 127) * 16;
  const _Float16* Qb = Q  + (size_t)bn * Sc * Hc;
  const _Float16* Kb = K  + (size_t)bn * Sc * Hc;
  const _Float16* Vb = Vt + (size_t)bn * Hc * Sc;
  _Float16*       Ob = attn + (size_t)bn * Sc * Hc;

  float mrow[8], lrow[8];
  v8f Oacc[16];
#pragma unroll
  for (int v = 0; v < 8; ++v) { mrow[v] = -1e30f; lrow[v] = 0.0f; }
#pragma unroll
  for (int ht = 0; ht < 16; ++ht)
#pragma unroll
    for (int v = 0; v < 8; ++v) Oacc[ht][v] = 0.0f;

  const int nkt = msk ? (((s0 + 15) >> 5) + 1) : (Sc >> 5);
  const float scale = 0.044194173824159216f;       // 1/sqrt(512)

  for (int kt = 0; kt < nkt; ++kt) {
    const int t0 = kt * 32;
    // ---- partial QK^T over this wave's 256-deep k-half ----
    v8f c0, c1;
#pragma unroll
    for (int v = 0; v < 8; ++v) { c0[v] = 0.0f; c1[v] = 0.0f; }
    for (int ks = 0; ks < 8; ++ks) {
      const int k0 = w * 256 + ks * 32;
      v16h a  = ld_afrag(Qb + (size_t)s0 * Hc + k0, Hc, lane);
      v16h b0 = ld_bfrag(Kb + (size_t)t0 * Hc + k0, Hc, lane);
      v16h b1 = ld_bfrag(Kb + (size_t)(t0 + 16) * Hc + k0, Hc, lane);
      c0 = wmma16(a, b0, c0);
      c1 = wmma16(a, b1, c1);
    }
    // ---- cross-wave reduction of the two k-halves via LDS ----
    {
      float* slab = &sx[w][0];
#pragma unroll
      for (int v = 0; v < 8; ++v) {
        const int row = v + 8 * g;
        slab[row * 32 + cn]      = c0[v];
        slab[row * 32 + 16 + cn] = c1[v];
      }
    }
    __syncthreads();
#pragma unroll
    for (int v = 0; v < 8; ++v) {
      const int row = v + 8 * g;
      c0[v] = sx[0][row * 32 + cn]      + sx[1][row * 32 + cn];
      c1[v] = sx[0][row * 32 + 16 + cn] + sx[1][row * 32 + 16 + cn];
    }
    __syncthreads();

    // ---- scale + causal mask ----
    const bool needmask = msk && (t0 + 31 > s0);
#pragma unroll
    for (int v = 0; v < 8; ++v) {
      const int srow = s0 + v + 8 * g;
      float x0 = c0[v] * scale;
      float x1 = c1[v] * scale;
      if (needmask) {
        if (t0 + cn > srow)      x0 = -1e30f;
        if (t0 + 16 + cn > srow) x1 = -1e30f;
      }
      c0[v] = x0; c1[v] = x1;
    }

    // ---- online softmax (rows live across lanes 0..15 of each half) ----
    float alpha[8];
    _Float16* ptw = &pt[w][0];
#pragma unroll
    for (int v = 0; v < 8; ++v) {
      float t = fmaxf(c0[v], c1[v]);
      t = fmaxf(t, __shfl_xor(t, 1, 32));
      t = fmaxf(t, __shfl_xor(t, 2, 32));
      t = fmaxf(t, __shfl_xor(t, 4, 32));
      t = fmaxf(t, __shfl_xor(t, 8, 32));
      const float mnew = fmaxf(mrow[v], t);
      alpha[v] = __expf(mrow[v] - mnew);
      mrow[v]  = mnew;
      const float e0 = __expf(c0[v] - mnew);
      const float e1 = __expf(c1[v] - mnew);
      const int row = v + 8 * g;
      ptw[row * 40 + cn]      = (_Float16)e0;     // C-layout -> LDS
      ptw[row * 40 + 16 + cn] = (_Float16)e1;
      float s = e0 + e1;
      s += __shfl_xor(s, 1, 32);
      s += __shfl_xor(s, 2, 32);
      s += __shfl_xor(s, 4, 32);
      s += __shfl_xor(s, 8, 32);
      lrow[v] = lrow[v] * alpha[v] + s;
    }
    asm volatile("s_wait_dscnt 0" ::: "memory");   // same-wave LDS RAW
    // P as A-fragment (LDS transpose of C-layout)
    v16h pa = ld_pair(ptw + cn * 40 + g * 8, ptw + cn * 40 + 16 + g * 8);

    // ---- P*V for this wave's 256-wide h-half ----
#pragma unroll
    for (int ht = 0; ht < 16; ++ht) {
#pragma unroll
      for (int v = 0; v < 8; ++v) Oacc[ht][v] *= alpha[v];
      const int h0 = w * 256 + ht * 16;
      v16h b = ld_bfrag(Vb + (size_t)h0 * Sc + t0, Sc, lane);
      Oacc[ht] = wmma16(pa, b, Oacc[ht]);
    }
  }

  // ---- normalize and store f16 attention output ----
  float inv[8];
#pragma unroll
  for (int v = 0; v < 8; ++v) inv[v] = 1.0f / lrow[v];
#pragma unroll
  for (int ht = 0; ht < 16; ++ht) {
    const int h0 = w * 256 + ht * 16;
#pragma unroll
    for (int v = 0; v < 8; ++v)
      Ob[(size_t)(s0 + v + 8 * g) * Hc + h0 + cn] = (_Float16)(Oacc[ht][v] * inv[v]);
  }
}

// ---------------- kernel 4: final projection (TDM-staged weights) ----------------
// Block = 8 waves computing a 128x64 output tile. The shared 64(N)x128(K) weight
// chunk is DMA'd into LDS by the Tensor Data Mover (double buffered, TENSORcnt
// pipelined); TDM pads 8 halves per 128-half row -> LDS stride 136 (bank-safe).
__global__ __launch_bounds__(256) void final_gemm(
    const _Float16* __restrict__ attn, const _Float16* __restrict__ wf,
    const float* __restrict__ bf, float* __restrict__ out) {
  constexpr int CHK  = 128;              // K chunk (halves)
  constexpr int LDB  = CHK + 8;          // padded LDS row stride
  constexpr int NC   = Ec / CHK;         // 8 chunks
  __shared__ __align__(16) _Float16 btile[2][64 * LDB];

  const int lane = threadIdx.x & 31;
  const int w    = threadIdx.x >> 5;
  const int rbB  = blockIdx.x >> 4;      // 0..63 (128-row block)
  const int cg   = blockIdx.x & 15;      // 0..15 (64-col group)
  const int r0   = rbB * 128 + w * 16;
  const int n0   = cg * 64;
  const int cn   = lane & 15, g = lane >> 4;

  // per-lane A row (token) split over the two head planes
  const int r  = r0 + cn;
  const int bb = r >> 11;                // / S
  const int s  = r & (Sc - 1);
  const _Float16* row0 = attn + ((size_t)(bb * 2 + 0) * Sc + s) * Hc;
  const _Float16* row1 = attn + ((size_t)(bb * 2 + 1) * Sc + s) * Hc;

  v8f acc[4];
#pragma unroll
  for (int j = 0; j < 4; ++j) {
    float bval = bf[n0 + j * 16 + cn];
#pragma unroll
    for (int v = 0; v < 8; ++v) acc[j][v] = bval;
  }

  // prologue: wave 0 issues TDM for chunk 0
  if (threadIdx.x < 32) {
    tdm_load_2d((unsigned)(uintptr_t)&btile[0][0], wf + (size_t)n0 * Ec,
                Ec, Ec, CHK, 64);
  }

  for (int c = 0; c < NC; ++c) {
    __syncthreads();                     // everyone done reading buf[(c+1)&1]
    if (threadIdx.x < 32) {
      if (c + 1 < NC) {
        tdm_load_2d((unsigned)(uintptr_t)&btile[(c + 1) & 1][0],
                    wf + (size_t)n0 * Ec + (c + 1) * CHK, Ec, Ec, CHK, 64);
        __builtin_amdgcn_s_wait_tensorcnt(1);   // oldest (chunk c) complete
      } else {
        __builtin_amdgcn_s_wait_tensorcnt(0);
      }
    }
    __syncthreads();                     // publish chunk c to all waves

    const _Float16* Bl = &btile[c & 1][0];
#pragma unroll
    for (int ks = 0; ks < CHK / 32; ++ks) {
      const int k0g = c * CHK + ks * 32;
      const int kk0 = k0g + g * 8;
      const int kk1 = k0g + 16 + g * 8;
      const _Float16* p0 = ((kk0 < Hc) ? row0 : row1) + (kk0 & (Hc - 1));
      const _Float16* p1 = ((kk1 < Hc) ? row0 : row1) + (kk1 & (Hc - 1));
      v16h a = ld_pair(p0, p1);
#pragma unroll
      for (int j = 0; j < 4; ++j) {
        v16h b = ld_bfrag(Bl + (size_t)(j * 16) * LDB + ks * 32, LDB, lane);
        acc[j] = wmma16(a, b, acc[j]);
      }
    }
  }

#pragma unroll
  for (int j = 0; j < 4; ++j)
#pragma unroll
    for (int v = 0; v < 8; ++v)
      out[(size_t)(r0 + v + 8 * g) * Ec + n0 + j * 16 + cn] = acc[j][v];
}

// ---------------- launcher ----------------
extern "C" void kernel_launch(void* const* d_in, const int* in_sizes, int n_in,
                              void* d_out, int out_size, void* d_ws, size_t ws_size,
                              hipStream_t stream) {
  const float* in  = (const float*)d_in[0];
  const float* Wq  = (const float*)d_in[1];
  const float* bq  = (const float*)d_in[2];
  const float* Wk  = (const float*)d_in[3];
  const float* bk  = (const float*)d_in[4];
  const float* Wv  = (const float*)d_in[5];
  const float* bv  = (const float*)d_in[6];
  const float* Wf  = (const float*)d_in[7];
  const float* bf  = (const float*)d_in[8];
  const int* masked = (const int*)d_in[9];

  _Float16* ws = (_Float16*)d_ws;
  _Float16* xh  = ws + XH_OFF;
  _Float16* Qp  = ws + Q_OFF;
  _Float16* Kp  = ws + K_OFF;
  _Float16* Vtp = ws + VT_OFF;
  _Float16* At  = ws + AT_OFF;
  _Float16* wq16 = ws + WQ_OFF;
  _Float16* wk16 = ws + WK_OFF;
  _Float16* wv16 = ws + WV_OFF;
  _Float16* wf16 = ws + WF_OFF;

  // 1) precision conversion / layout
  cast_f16<<<(int)(W_ELE / 256), 256, 0, stream>>>(Wq, wq16, (int)W_ELE);
  cast_f16<<<(int)(W_ELE / 256), 256, 0, stream>>>(Wk, wk16, (int)W_ELE);
  cast_f16<<<(int)(W_ELE / 256), 256, 0, stream>>>(Wv, wv16, (int)W_ELE);
  cast_f16<<<(int)((size_t)Ec * Ec / 256), 256, 0, stream>>>(Wf, wf16, Ec * Ec);
  remap_input<<<(int)(PLANE / 256), 256, 0, stream>>>(in, xh);

  // 2) Q,K,V projections (WMMA)
  qkv_gemm<<<dim3(1024, 3), 256, 0, stream>>>(xh, wq16, wk16, wv16, bq, bk, bv,
                                              Qp, Kp, Vtp);

  // 3) flash attention (WMMA QK^T + WMMA P*V, online softmax)
  flash_attn<<<1024, 64, 0, stream>>>(Qp, Kp, Vtp, At, masked);

  // 4) output projection (WMMA, TDM-staged weights) -> fp32 result
  final_gemm<<<1024, 256, 0, stream>>>(At, wf16, bf, (float*)d_out);
}
